// teacher_38010460570242
// MI455X (gfx1250) — matrix operand order repaired and checked by reading, not compiled, not run on hardware
//
#include <hip/hip_runtime.h>
#include <hip/hip_bf16.h>

// ===========================================================================
// SAM ViT-B image encoder forward on gfx1250 (MI455X).
// All matmul-shaped work runs through v_wmma_f32_16x16x32_bf16 (wave32 WMMA).
// GEMM: 128x128 block tile, BK=32, 8 waves (4x2), wave tile 32x64 (2x4 frags),
// float4 global staging, register prefetch + double-buffered LDS (1 barrier
// per K-tile), bf16 conversion via compiler-selected native lowering.
// ===========================================================================

typedef __attribute__((ext_vector_type(16))) __bf16 v16bf;
typedef __attribute__((ext_vector_type(2)))  __bf16 v2bf;
typedef __attribute__((ext_vector_type(8)))  float  v8f;

#define BM 128
#define BN 128
#define BKK 32
#define LDSS 40   // ushorts per LDS row (32 + 8 pad) = 80B: 16B-aligned rows
#define LDSBUF (BM * LDSS)   // one double-buffer plane (5120 ushorts = 10240B)

#define F_BTRANS 1
#define F_BIAS   2
#define F_GELU   4

// Let clang lower float->bf16 (RNE); emits v_cvt_pk_bf16_f32 when the target
// has it, software RNE otherwise.
__device__ __forceinline__ unsigned int pack2(float a, float b) {
  v2bf v;
  v[0] = (__bf16)a;
  v[1] = (__bf16)b;
  union { v2bf v; unsigned int u; } cv;
  cv.v = v;
  return cv.u;
}
__device__ __forceinline__ unsigned short f2bf(float f) {
  union { __bf16 h; unsigned short s; } cv;
  cv.h = (__bf16)f;
  return cv.s;
}

// ---------------------------------------------------------------------------
// Generic batched GEMM: C = alpha * A(MxK) * op(B) + bias, optional GELU.
//   flags & F_BTRANS : B is W[N,K] row-major (C = A @ W^T)   -- weight form
//   else             : B is [K,N] row-major (transposed while staging to LDS)
// ---------------------------------------------------------------------------
__global__ __launch_bounds__(256)
void gemm_wmma(const float* __restrict__ A, const float* __restrict__ B,
               const float* __restrict__ bias, float* __restrict__ C,
               int M, int N, int K, int lda, int ldb, int ldc,
               long long aOut, long long aIn, long long bOut, long long bIn,
               long long cOut, long long cIn, int innerCnt,
               int flags, float alpha)
{
  __shared__ __align__(16) unsigned short As[2 * LDSBUF];
  __shared__ __align__(16) unsigned short Bs[2 * LDSBUF];

  const int z  = blockIdx.z;
  const int zo = z / innerCnt, zi = z % innerCnt;
  A += zo * aOut + zi * aIn;
  B += zo * bOut + zi * bIn;
  C += zo * cOut + zi * cIn;

  const int tid  = threadIdx.x;
  const int wave = tid >> 5;
  const int lane = tid & 31;
  const int wm = wave & 3;          // 4 wave rows
  const int wn = wave >> 2;         // 2 wave cols
  const int rowBase = blockIdx.y * BM;
  const int colBase = blockIdx.x * BN;
  const int laneHalf = lane >> 4;   // 0..1
  const int laneMod  = lane & 15;

  const bool btrans = (flags & F_BTRANS) != 0;
  const bool rowsFullA = (rowBase + BM <= M);
  const bool colsFullB = (colBase + BN <= N);

  // staging maps (float4 granularity)
  const int sAr = tid >> 3;         // 0..31, +32 per iter (A / B-trans tiles)
  const int sAc = (tid & 7) * 4;    // k offset 0..28
  const int sBk = tid >> 5;         // 0..7, +8 per iter (B non-trans tiles)
  const int sBc = (tid & 31) * 4;   // n offset 0..124

  float4 aReg[4], bReg[4];

  auto loadAB = [&](int kb) {
    const bool kFull = (kb + BKK <= K);
    // ---- A: BM x BKK ----
    if (rowsFullA & kFull) {
      #pragma unroll
      for (int i = 0; i < 4; ++i)
        aReg[i] = *(const float4*)(A + (long long)(rowBase + sAr + i * 32) * lda
                                     + (kb + sAc));
    } else {
      #pragma unroll
      for (int i = 0; i < 4; ++i) {
        int gr = rowBase + sAr + i * 32;
        bool rok = gr < M;
        const float* base = A + (long long)(rok ? gr : 0) * lda;
        int k0 = kb + sAc;
        float t0 = base[min(k0,     K - 1)];
        float t1 = base[min(k0 + 1, K - 1)];
        float t2 = base[min(k0 + 2, K - 1)];
        float t3 = base[min(k0 + 3, K - 1)];
        float4 v;
        v.x = (rok && k0     < K) ? t0 : 0.f;
        v.y = (rok && k0 + 1 < K) ? t1 : 0.f;
        v.z = (rok && k0 + 2 < K) ? t2 : 0.f;
        v.w = (rok && k0 + 3 < K) ? t3 : 0.f;
        aReg[i] = v;
      }
    }
    // ---- B ----
    if (btrans) {                          // W[N,K]: same shape as A staging
      if (colsFullB & kFull) {
        #pragma unroll
        for (int i = 0; i < 4; ++i)
          bReg[i] = *(const float4*)(B + (long long)(colBase + sAr + i * 32) * ldb
                                       + (kb + sAc));
      } else {
        #pragma unroll
        for (int i = 0; i < 4; ++i) {
          int gn = colBase + sAr + i * 32;
          bool nok = gn < N;
          const float* base = B + (long long)(nok ? gn : 0) * ldb;
          int k0 = kb + sAc;
          float t0 = base[min(k0,     K - 1)];
          float t1 = base[min(k0 + 1, K - 1)];
          float t2 = base[min(k0 + 2, K - 1)];
          float t3 = base[min(k0 + 3, K - 1)];
          float4 v;
          v.x = (nok && k0     < K) ? t0 : 0.f;
          v.y = (nok && k0 + 1 < K) ? t1 : 0.f;
          v.z = (nok && k0 + 2 < K) ? t2 : 0.f;
          v.w = (nok && k0 + 3 < K) ? t3 : 0.f;
          bReg[i] = v;
        }
      }
    } else {                               // B[K,N]: BKK x BN tile
      if (colsFullB & kFull) {
        #pragma unroll
        for (int i = 0; i < 4; ++i)
          bReg[i] = *(const float4*)(B + (long long)(kb + sBk + i * 8) * ldb
                                       + (colBase + sBc));
      } else {
        #pragma unroll
        for (int i = 0; i < 4; ++i) {
          int gk = kb + sBk + i * 8;
          bool kok = gk < K;
          const float* base = B + (long long)(kok ? gk : 0) * ldb;
          int c0 = colBase + sBc;
          float t0 = base[min(c0,     N - 1)];
          float t1 = base[min(c0 + 1, N - 1)];
          float t2 = base[min(c0 + 2, N - 1)];
          float t3 = base[min(c0 + 3, N - 1)];
          float4 v;
          v.x = (kok && c0     < N) ? t0 : 0.f;
          v.y = (kok && c0 + 1 < N) ? t1 : 0.f;
          v.z = (kok && c0 + 2 < N) ? t2 : 0.f;
          v.w = (kok && c0 + 3 < N) ? t3 : 0.f;
          bReg[i] = v;
        }
      }
    }
  };

  auto storeAB = [&](int buf) {
    unsigned short* as = As + buf * LDSBUF;
    unsigned short* bs = Bs + buf * LDSBUF;
    #pragma unroll
    for (int i = 0; i < 4; ++i) {
      uint2 pk; pk.x = pack2(aReg[i].x, aReg[i].y);
      pk.y = pack2(aReg[i].z, aReg[i].w);
      *(uint2*)&as[(sAr + i * 32) * LDSS + sAc] = pk;
    }
    if (btrans) {
      #pragma unroll
      for (int i = 0; i < 4; ++i) {
        uint2 pk; pk.x = pack2(bReg[i].x, bReg[i].y);
        pk.y = pack2(bReg[i].z, bReg[i].w);
        *(uint2*)&bs[(sAr + i * 32) * LDSS + sAc] = pk;
      }
    } else {                               // transpose into Bs[n][k]
      #pragma unroll
      for (int i = 0; i < 4; ++i) {
        int kr = sBk + i * 8;
        bs[(sBc + 0) * LDSS + kr] = f2bf(bReg[i].x);
        bs[(sBc + 1) * LDSS + kr] = f2bf(bReg[i].y);
        bs[(sBc + 2) * LDSS + kr] = f2bf(bReg[i].z);
        bs[(sBc + 3) * LDSS + kr] = f2bf(bReg[i].w);
      }
    }
  };

  v8f acc[2][4] = {};

  loadAB(0);
  storeAB(0);
  __syncthreads();
  int cur = 0;
  for (int kb = 0;;) {
    int kn = kb + BKK;
    if (kn < K) loadAB(kn);   // prefetch next tile: overlaps WMMA below

    // ---- fragment loads (ISA 16-bit A 16x32 / B 32x16 layouts) ----
    const unsigned short* as = As + cur * LDSBUF;
    const unsigned short* bs = Bs + cur * LDSBUF;
    v16bf afrag[2], bfrag[4];
    #pragma unroll
    for (int fm = 0; fm < 2; ++fm) {
      int r = wm * 32 + fm * 16 + laneMod;
      const uint4* p = (const uint4*)&as[r * LDSS + laneHalf * 8];
      union { v16bf v; uint4 q[2]; } u;
      u.q[0] = p[0];     // K = off .. off+7
      u.q[1] = p[2];     // K = off+16 .. off+23
      afrag[fm] = u.v;
    }
    #pragma unroll
    for (int fn = 0; fn < 4; ++fn) {
      int r = wn * 64 + fn * 16 + laneMod;
      const uint4* p = (const uint4*)&bs[r * LDSS + laneHalf * 8];
      union { v16bf v; uint4 q[2]; } u;
      u.q[0] = p[0];     // K = half*8 .. +7
      u.q[1] = p[2];     // K = 16+half*8 .. +7
      bfrag[fn] = u.v;
    }
    #pragma unroll
    for (int fm = 0; fm < 2; ++fm)
      #pragma unroll
      for (int fn = 0; fn < 4; ++fn)
        acc[fm][fn] = __builtin_amdgcn_wmma_f32_16x16x32_bf16(
            false, afrag[fm], false, bfrag[fn], (short)0, acc[fm][fn],
            false, false);

    if (kn >= K) break;
    storeAB(cur ^ 1);         // write the other plane while others still read
    __syncthreads();          // single barrier per K-tile
    cur ^= 1;
    kb = kn;
  }

  // ---- epilogue: C/D layout lanes0-15 -> M=r, lanes16-31 -> M=r+8 ----
  const bool hasBias = (flags & F_BIAS) != 0;
  const bool doGelu  = (flags & F_GELU) != 0;
  const bool fullC   = rowsFullA && colsFullB;
  #pragma unroll
  for (int fm = 0; fm < 2; ++fm)
    #pragma unroll
    for (int fn = 0; fn < 4; ++fn) {
      int col = colBase + wn * 64 + fn * 16 + laneMod;
      float bv = (hasBias && (fullC || col < N)) ? bias[col] : 0.f;
      #pragma unroll
      for (int r = 0; r < 8; ++r) {
        int row = rowBase + wm * 32 + fm * 16 + r + laneHalf * 8;
        if (fullC || (row < M && col < N)) {
          float v = acc[fm][fn][r] * alpha + bv;
          if (doGelu) v = 0.5f * v * (1.f + erff(v * 0.70710678118654752f));
          C[(long long)row * ldc + col] = v;
        }
      }
    }
}

// ---------------------------------------------------------------------------
// LayerNorm over last dim D (eps 1e-6); one 256-thread block per row.
// ---------------------------------------------------------------------------
__global__ __launch_bounds__(256)
void ln_kernel(const float* __restrict__ in, float* __restrict__ out,
               const float* __restrict__ w, const float* __restrict__ b, int D)
{
  long long row = blockIdx.x;
  const float* x = in + row * D;
  float* y = out + row * D;
  __shared__ float red[256];
  int tid = threadIdx.x;
  float s = 0.f;
  for (int i = tid; i < D; i += 256) s += x[i];
  red[tid] = s; __syncthreads();
  for (int st = 128; st > 0; st >>= 1) {
    if (tid < st) red[tid] += red[tid + st];
    __syncthreads();
  }
  float mean = red[0] / (float)D; __syncthreads();
  float v = 0.f;
  for (int i = tid; i < D; i += 256) { float d = x[i] - mean; v += d * d; }
  red[tid] = v; __syncthreads();
  for (int st = 128; st > 0; st >>= 1) {
    if (tid < st) red[tid] += red[tid + st];
    __syncthreads();
  }
  float rstd = rsqrtf(red[0] / (float)D + 1e-6f);
  for (int i = tid; i < D; i += 256)
    y[i] = (x[i] - mean) * rstd * w[i] + b[i];
}

// ---------------------------------------------------------------------------
// Elementwise helpers
// ---------------------------------------------------------------------------
__global__ void add_inplace(float* __restrict__ X, const float* __restrict__ Y,
                            long long n) {
  long long i = (long long)blockIdx.x * 256 + threadIdx.x;
  if (i < n) X[i] += Y[i];
}

// patch im2col: P[t=gh*64+gw][k=c*256+i*16+j] = x[c,gh*16+i,gw*16+j]
__global__ void im2col_patch(const float* __restrict__ x, float* __restrict__ P) {
  long long idx = (long long)blockIdx.x * 256 + threadIdx.x;
  if (idx >= (long long)4096 * 768) return;
  int k = (int)(idx % 768);
  int t = (int)(idx / 768);
  int gh = t >> 6, gw = t & 63;
  int c = k >> 8, rem = k & 255, i = rem >> 4, j = rem & 15;
  P[idx] = x[(long long)c * 1048576 + (long long)(gh * 16 + i) * 1024 + gw * 16 + j];
}

// 3x3 SAME im2col on [64,64,256]; K order = i*9 + dy*3 + dx (matches w2[o][i][dy][dx])
__global__ void im2col3(const float* __restrict__ Y1, float* __restrict__ P2) {
  long long idx = (long long)blockIdx.x * 256 + threadIdx.x;
  if (idx >= (long long)4096 * 2304) return;
  int k = (int)(idx % 2304);
  int t = (int)(idx / 2304);
  int y = t >> 6, x = t & 63;
  int i = k / 9, rem = k % 9, dy = rem / 3, dx = rem % 3;
  int yy = y + dy - 1, xx = x + dx - 1;
  float v = 0.f;
  if (yy >= 0 && yy < 64 && xx >= 0 && xx < 64)
    v = Y1[(long long)(yy * 64 + xx) * 256 + i];
  P2[idx] = v;
}

// window gather: WQKV[25][208][2304]; pads(grid 64->70) get qkv_b, rows 196..207 get 0
__global__ void gather_wqkv(const float* __restrict__ QKV,
                            const float* __restrict__ qkv_b,
                            float* __restrict__ WQKV) {
  long long idx = (long long)blockIdx.x * 256 + threadIdx.x;
  if (idx >= (long long)25 * 208 * 2304) return;
  int c = (int)(idx % 2304);
  long long r = idx / 2304;
  int t = (int)(r % 208);
  int win = (int)(r / 208);
  int wy = win / 5, wx = win % 5;
  float v = 0.f;
  if (t < 196) {
    int iy = t / 14, ix = t % 14;
    int y = wy * 14 + iy, x = wx * 14 + ix;
    if (y < 64 && x < 64) v = QKV[(long long)(y * 64 + x) * 2304 + c];
    else                  v = qkv_b[c];     // zero-padded token -> qkv == bias
  }
  WQKV[idx] = v;
}

// window scatter: AO[4096][768] <- WO[25][208][768]
__global__ void scatter_win(const float* __restrict__ WO, float* __restrict__ AO) {
  long long idx = (long long)blockIdx.x * 256 + threadIdx.x;
  if (idx >= (long long)4096 * 768) return;
  int c = (int)(idx % 768);
  int t = (int)(idx / 768);
  int y = t >> 6, x = t & 63;
  int win = (y / 14) * 5 + (x / 14);
  int wt = (y % 14) * 14 + (x % 14);
  AO[idx] = WO[(long long)win * (208 * 768) + (long long)wt * 768 + c];
}

__device__ __forceinline__ void dot2_64(const float* __restrict__ qv,
                                        const float* __restrict__ Rh,
                                        const float* __restrict__ Rw,
                                        float& sh, float& sw) {
  const float4* q4 = (const float4*)qv;
  const float4* h4 = (const float4*)Rh;
  const float4* w4 = (const float4*)Rw;
  float a = 0.f, b = 0.f;
  #pragma unroll
  for (int c = 0; c < 16; ++c) {
    float4 q = q4[c], h = h4[c], w = w4[c];
    a += q.x * h.x + q.y * h.y + q.z * h.z + q.w * h.w;
    b += q.x * w.x + q.y * w.y + q.z * w.z + q.w * w.w;
  }
  sh = a; sw = b;
}

// rel-pos dots for windows: RH/RW [win*12+head][196][14]
__global__ void relpos_win(const float* __restrict__ WQKV,
                           const float* __restrict__ rel_h,
                           const float* __restrict__ rel_w,
                           float* __restrict__ RH, float* __restrict__ RW) {
  long long idx = (long long)blockIdx.x * 256 + threadIdx.x;
  if (idx >= (long long)25 * 12 * 196 * 14) return;
  int kh = (int)(idx % 14);
  long long r = idx / 14;
  int q = (int)(r % 196);
  long long bh = r / 196;
  int head = (int)(bh % 12);
  int win  = (int)(bh / 12);
  const float* qv = WQKV + (long long)win * 479232 + (long long)q * 2304 + head * 64;
  int iy = q / 14, ix = q % 14;
  float sh, sw;
  dot2_64(qv, rel_h + (long long)(iy - kh + 13) * 64,
              rel_w + (long long)(ix - kh + 13) * 64, sh, sw);
  RH[idx] = sh; RW[idx] = sw;
}

// rel-pos dots for global attention: RH/RW [head][4096][64]
__global__ void relpos_glob(const float* __restrict__ QKV,
                            const float* __restrict__ rel_h,
                            const float* __restrict__ rel_w,
                            float* __restrict__ RH, float* __restrict__ RW) {
  long long idx = (long long)blockIdx.x * 256 + threadIdx.x;
  if (idx >= (long long)12 * 4096 * 64) return;
  int kh = (int)(idx & 63);
  long long r = idx >> 6;
  int q = (int)(r & 4095);
  int head = (int)(r >> 12);
  const float* qv = QKV + (long long)q * 2304 + head * 64;
  int yq = q >> 6, xq = q & 63;
  float sh, sw;
  dot2_64(qv, rel_h + (long long)(yq - kh + 63) * 64,
              rel_w + (long long)(xq - kh + 63) * 64, sh, sw);
  RH[idx] = sh; RW[idx] = sw;
}

// windowed softmax: rows of S[300][208][208]; valid q<196, valid k<196
__global__ __launch_bounds__(256)
void softmax_win(float* __restrict__ S, const float* __restrict__ RH,
                 const float* __restrict__ RW) {
  int rowId = blockIdx.x;                 // 300*208 rows
  int q = rowId % 208;
  int bh = rowId / 208;
  float* row = S + (long long)bh * 43264 + (long long)q * 208;
  int tid = threadIdx.x;
  if (q >= 196) { if (tid < 208) row[tid] = 0.f; return; }
  __shared__ float red[256];
  const float* rh = RH + ((long long)bh * 196 + q) * 14;
  const float* rw = RW + ((long long)bh * 196 + q) * 14;
  float sv = 0.f, v = -1e30f;
  if (tid < 196) { sv = row[tid] + rh[tid / 14] + rw[tid % 14]; v = sv; }
  red[tid] = v; __syncthreads();
  for (int st = 128; st > 0; st >>= 1) {
    if (tid < st) red[tid] = fmaxf(red[tid], red[tid + st]);
    __syncthreads();
  }
  float m = red[0]; __syncthreads();
  float e = (tid < 196) ? __expf(sv - m) : 0.f;
  red[tid] = e; __syncthreads();
  for (int st = 128; st > 0; st >>= 1) {
    if (tid < st) red[tid] += red[tid + st];
    __syncthreads();
  }
  float inv = 1.f / red[0];
  if (tid < 208) row[tid] = (tid < 196) ? e * inv : 0.f;
}

// global softmax: rows of S[4096][4096] for one head; bias via RH/RW [q][64]
__global__ __launch_bounds__(256)
void softmax_glob(float* __restrict__ S, const float* __restrict__ RH,
                  const float* __restrict__ RW) {
  int q = blockIdx.x;
  float* row = S + (long long)q * 4096;
  const float* rh = RH + (long long)q * 64;
  const float* rw = RW + (long long)q * 64;
  int tid = threadIdx.x;
  __shared__ float red[256];
  float loc[16];
  float m = -1e30f;
  #pragma unroll
  for (int i = 0; i < 16; ++i) {
    int k = tid + i * 256;
    float sv = row[k] + rh[k >> 6] + rw[k & 63];
    loc[i] = sv; m = fmaxf(m, sv);
  }
  red[tid] = m; __syncthreads();
  for (int st = 128; st > 0; st >>= 1) {
    if (tid < st) red[tid] = fmaxf(red[tid], red[tid + st]);
    __syncthreads();
  }
  m = red[0]; __syncthreads();
  float s = 0.f;
  #pragma unroll
  for (int i = 0; i < 16; ++i) { loc[i] = __expf(loc[i] - m); s += loc[i]; }
  red[tid] = s; __syncthreads();
  for (int st = 128; st > 0; st >>= 1) {
    if (tid < st) red[tid] += red[tid + st];
    __syncthreads();
  }
  float inv = 1.f / red[0];
  #pragma unroll
  for (int i = 0; i < 16; ++i) row[tid + i * 256] = loc[i] * inv;
}

// final NCHW transpose: out[c][y][x] = Y[(y*64+x)*256 + c]
__global__ void nchw_out(const float* __restrict__ Y, float* __restrict__ out) {
  long long idx = (long long)blockIdx.x * 256 + threadIdx.x;
  if (idx >= (long long)256 * 4096) return;
  int c = (int)(idx / 4096);
  int t = (int)(idx % 4096);
  out[idx] = Y[(long long)t * 256 + c];
}

// ---------------------------------------------------------------------------
// Host side
// ---------------------------------------------------------------------------
static inline void launch_gemm(hipStream_t s, const float* A, const float* B,
                               const float* bias, float* C,
                               int M, int N, int K, int lda, int ldb, int ldc,
                               long long aO, long long aI, long long bO, long long bI,
                               long long cO, long long cI, int inner, int batch,
                               int flags, float alpha) {
  dim3 grid((N + BN - 1) / BN, (M + BM - 1) / BM, batch);
  gemm_wmma<<<grid, 256, 0, s>>>(A, B, bias, C, M, N, K, lda, ldb, ldc,
                                 aO, aI, bO, bI, cO, cI, inner, flags, alpha);
}

static inline long long ew_grid(long long n) { return (n + 255) / 256; }

extern "C" void kernel_launch(void* const* d_in, const int* in_sizes, int n_in,
                              void* d_out, int out_size, void* d_ws, size_t ws_size,
                              hipStream_t stream) {
  (void)in_sizes; (void)n_in; (void)out_size;

  // ---- input map (setup_inputs dict order, depth-first) ----
  const float* xin     = (const float*)d_in[0];   // [3,1024,1024]
  const float* patch_w = (const float*)d_in[1];   // [768,768]
  const float* patch_b = (const float*)d_in[2];
  const float* pos     = (const float*)d_in[3];   // [4096,768]
  const float* neck_w1 = (const float*)d_in[172]; // [256,768]
  const float* ln1_w   = (const float*)d_in[173];
  const float* ln1_b   = (const float*)d_in[174];
  const float* neck_w2 = (const float*)d_in[175]; // [256,2304]
  const float* ln2_w   = (const float*)d_in[176];
  const float* ln2_b   = (const float*)d_in[177];

  // ---- workspace bump allocation (bytes) ----
  size_t off = 0;
  auto alloc = [&](size_t bytes) { size_t r = off; off += (bytes + 255) & ~(size_t)255; return r; };
  size_t oX    = alloc(4096ull * 768 * 4);
  size_t oXN   = alloc(4096ull * 768 * 4);
  size_t oQKV  = alloc(4096ull * 2304 * 4);
  size_t oAO   = alloc(4096ull * 768 * 4);
  size_t oY    = alloc(4096ull * 768 * 4);
  size_t oH1   = alloc(4096ull * 3072 * 4);
  size_t oWQKV = alloc(25ull * 208 * 2304 * 4);
  size_t oS    = alloc(4096ull * 4096 * 4);       // covers windowed 300*208*208 too
  size_t oWO   = alloc(25ull * 208 * 768 * 4);
  size_t oRH   = alloc(12ull * 4096 * 64 * 4);    // covers windowed rel bias too
  size_t oRW   = alloc(12ull * 4096 * 64 * 4);
  if (ws_size < off) return;   // workspace too small: bail (no OOB writes)

  char* ws = (char*)d_ws;
  float* X    = (float*)(ws + oX);
  float* XN   = (float*)(ws + oXN);
  float* QKV  = (float*)(ws + oQKV);
  float* AO   = (float*)(ws + oAO);
  float* Y    = (float*)(ws + oY);
  float* H1   = (float*)(ws + oH1);
  float* WQKV = (float*)(ws + oWQKV);
  float* S    = (float*)(ws + oS);
  float* WO   = (float*)(ws + oWO);
  float* RH   = (float*)(ws + oRH);
  float* RW   = (float*)(ws + oRW);
  float* P    = H1;    // patch im2col reuses MLP scratch (pre-blocks)
  float* P2   = QKV;   // neck im2col reuses QKV scratch (post-blocks)

  // ---- patch embed: im2col + GEMM (+bias), then + pos_embed ----
  im2col_patch<<<ew_grid(4096ll * 768), 256, 0, stream>>>(xin, P);
  launch_gemm(stream, P, patch_w, patch_b, X, 4096, 768, 768, 768, 768, 768,
              0, 0, 0, 0, 0, 0, 1, 1, F_BTRANS | F_BIAS, 1.f);
  add_inplace<<<ew_grid(4096ll * 768), 256, 0, stream>>>(X, pos, 4096ll * 768);

  // ---- transformer blocks ----
  for (int b = 0; b < 12; ++b) {
    const float* const* bp = (const float* const*)&d_in[4 + b * 14];
    const float* n1_w = bp[0], *n1_b = bp[1];
    const float* qkv_w = bp[2], *qkv_b = bp[3];
    const float* proj_w = bp[4], *proj_b = bp[5];
    const float* rel_h = bp[6], *rel_w = bp[7];
    const float* n2_w = bp[8], *n2_b = bp[9];
    const float* m1_w = bp[10], *m1_b = bp[11];
    const float* m2_w = bp[12], *m2_b = bp[13];
    const bool isGlobal = (b == 2 || b == 5 || b == 8 || b == 11);

    ln_kernel<<<4096, 256, 0, stream>>>(X, XN, n1_w, n1_b, 768);
    launch_gemm(stream, XN, qkv_w, qkv_b, QKV, 4096, 2304, 768, 768, 768, 2304,
                0, 0, 0, 0, 0, 0, 1, 1, F_BTRANS | F_BIAS, 1.f);

    if (!isGlobal) {
      gather_wqkv<<<ew_grid(25ll * 208 * 2304), 256, 0, stream>>>(QKV, qkv_b, WQKV);
      relpos_win<<<ew_grid(25ll * 12 * 196 * 14), 256, 0, stream>>>(
          WQKV, rel_h, rel_w, RH, RW);
      // scores[win,head] (208x208) = 0.125 * q @ k^T   (batched, 300 batches)
      launch_gemm(stream, WQKV, WQKV + 768, nullptr, S, 208, 208, 64,
                  2304, 2304, 208,
                  479232, 64, 479232, 64, 12ll * 43264, 43264,
                  12, 300, F_BTRANS, 0.125f);
      softmax_win<<<300 * 208, 256, 0, stream>>>(S, RH, RW);
      // out[win,head] (208x64) = probs @ v
      launch_gemm(stream, S, WQKV + 1536, nullptr, WO, 208, 64, 208,
                  208, 2304, 768,
                  12ll * 43264, 43264, 479232, 64, 159744, 64,
                  12, 300, 0, 1.f);
      scatter_win<<<ew_grid(4096ll * 768), 256, 0, stream>>>(WO, AO);
    } else {
      relpos_glob<<<ew_grid(12ll * 4096 * 64), 256, 0, stream>>>(
          QKV, rel_h, rel_w, RH, RW);
      for (int h = 0; h < 12; ++h) {   // 67MB score buffer reused per head
        launch_gemm(stream, QKV + h * 64, QKV + 768 + h * 64, nullptr, S,
                    4096, 4096, 64, 2304, 2304, 4096,
                    0, 0, 0, 0, 0, 0, 1, 1, F_BTRANS, 0.125f);
        softmax_glob<<<4096, 256, 0, stream>>>(
            S, RH + (long long)h * 4096 * 64, RW + (long long)h * 4096 * 64);
        launch_gemm(stream, S, QKV + 1536 + h * 64, nullptr, AO + h * 64,
                    4096, 64, 4096, 4096, 2304, 768,
                    0, 0, 0, 0, 0, 0, 1, 1, 0, 1.f);
      }
    }

    // proj + residual
    launch_gemm(stream, AO, proj_w, proj_b, Y, 4096, 768, 768, 768, 768, 768,
                0, 0, 0, 0, 0, 0, 1, 1, F_BTRANS | F_BIAS, 1.f);
    add_inplace<<<ew_grid(4096ll * 768), 256, 0, stream>>>(X, Y, 4096ll * 768);

    // MLP
    ln_kernel<<<4096, 256, 0, stream>>>(X, XN, n2_w, n2_b, 768);
    launch_gemm(stream, XN, m1_w, m1_b, H1, 4096, 3072, 768, 768, 768, 3072,
                0, 0, 0, 0, 0, 0, 1, 1, F_BTRANS | F_BIAS | F_GELU, 1.f);
    launch_gemm(stream, H1, m2_w, m2_b, Y, 4096, 768, 3072, 3072, 3072, 768,
                0, 0, 0, 0, 0, 0, 1, 1, F_BTRANS | F_BIAS, 1.f);
    add_inplace<<<ew_grid(4096ll * 768), 256, 0, stream>>>(X, Y, 4096ll * 768);
  }

  // ---- neck: 1x1 conv -> LN -> 3x3 conv -> LN -> NCHW ----
  launch_gemm(stream, X, neck_w1, nullptr, AO, 4096, 256, 768, 768, 768, 256,
              0, 0, 0, 0, 0, 0, 1, 1, F_BTRANS, 1.f);           // y1 in AO
  ln_kernel<<<4096, 256, 0, stream>>>(AO, Y, ln1_w, ln1_b, 256); // y1n in Y
  im2col3<<<ew_grid(4096ll * 2304), 256, 0, stream>>>(Y, P2);
  launch_gemm(stream, P2, neck_w2, nullptr, AO, 4096, 256, 2304, 2304, 2304, 256,
              0, 0, 0, 0, 0, 0, 1, 1, F_BTRANS, 1.f);           // y2 in AO
  ln_kernel<<<4096, 256, 0, stream>>>(AO, Y, ln2_w, ln2_b, 256); // y2n in Y
  nchw_out<<<ew_grid(256ll * 4096), 256, 0, stream>>>(Y, (float*)d_out);
}
